// DefectNetForceField_53334903882519
// MI455X (gfx1250) — matrix operand order repaired and checked by reading, not compiled, not run on hardware
//
#include <hip/hip_runtime.h>
#include <stdint.h>

typedef unsigned short u16;
typedef __attribute__((ext_vector_type(16))) __bf16 v16bf;
typedef __attribute__((ext_vector_type(8)))  float  v8f;
typedef __attribute__((ext_vector_type(4)))  unsigned int u32x4;

#define NN 50000
#define EE 1200000
#define TT 800000
#define BB 16
#define LL 4
#define KT2 7   /* ceil(208/32) K-tiles, 2-body */
#define KT3 8   /* ceil(240/32) K-tiles, 3-body */

__device__ inline u16 f2bf(float f) {
    unsigned int u = __float_as_uint(f);
    unsigned int r = u + 0x7FFFu + ((u >> 16) & 1u);   // round-to-nearest-even
    return (u16)(r >> 16);
}
__device__ inline float bf2f(u16 h) { return __uint_as_float(((unsigned int)h) << 16); }
__device__ inline float softplusf(float x) { return (x > 20.f) ? x : log1pf(expf(x)); }
__device__ inline float sigmoidf(float x) { return 1.f / (1.f + expf(-x)); }

union Frag16 { v16bf v; u32x4 q[2]; u16 us[16]; };
union FragF  { v8f v; float f[8]; };

// ---------------------------------------------------------------------------
// Weight pre-swizzle: f32 (L,Kreal,128) -> bf16 fragments [l][kt][nt][lane][16]
// fragment half j of lane L in (kt,nt): K = kt*32 + (L/16)*8 + (j&7) + (j>>3)*16, col = nt*16 + L%16
// (linear bias tb_b/th_b cancels inside BatchNorm, so it is dropped)
// ---------------------------------------------------------------------------
__global__ void prep_w_kernel(int Kreal, int ktiles, const float* __restrict__ W,
                              u16* __restrict__ out, long long total) {
    long long i = (long long)blockIdx.x * blockDim.x + threadIdx.x;
    if (i >= total) return;
    int j    = (int)(i & 15);
    int lane = (int)((i >> 4) & 31);
    int nt   = (int)((i >> 9) & 7);
    long long rest = i >> 12;
    int kt = (int)(rest % ktiles);
    int l  = (int)(rest / ktiles);
    int K   = kt * 32 + (lane >> 4) * 8 + (j & 7) + (j >> 3) * 16;
    int col = nt * 16 + (lane & 15);
    float v = (K < Kreal) ? W[((long long)l * Kreal + K) * 128 + col] : 0.f;
    out[i] = f2bf(v);
}

// ---------------------------------------------------------------------------
// Edge geometry: r_ij, dist, cosine cutoff, Gaussian distance basis (bf16)
// ---------------------------------------------------------------------------
__global__ void edge_geom_kernel(const int* __restrict__ ei, const float* __restrict__ pos,
                                 const float* __restrict__ eoff,
                                 float* __restrict__ r_ij, float* __restrict__ dist,
                                 float* __restrict__ cw, u16* __restrict__ ef_bf) {
    int e = blockIdx.x * blockDim.x + threadIdx.x;
    if (e >= EE) return;
    int s = ei[e], d = ei[EE + e];
    float rx = pos[s*3+0] + eoff[e*3+0] - pos[d*3+0];
    float ry = pos[s*3+1] + eoff[e*3+1] - pos[d*3+1];
    float rz = pos[s*3+2] + eoff[e*3+2] - pos[d*3+2];
    float dd = sqrtf(rx*rx + ry*ry + rz*rz);
    r_ij[e*3+0] = rx; r_ij[e*3+1] = ry; r_ij[e*3+2] = rz;
    dist[e] = dd;
    cw[e] = 0.5f * (cosf(dd * 3.14159265358979323846f / 5.0f) + 1.0f);
    const float inv = 1.f / (0.2f * 0.2f);
    u16* row = ef_bf + (size_t)e * 80;
    #pragma unroll 8
    for (int g = 0; g < 80; ++g) {
        float c = 5.0f * (float)g / 79.0f;
        float t = dd - c;
        row[g] = f2bf(expf(-t * t * inv));
    }
}

// ---------------------------------------------------------------------------
// Triplet geometry: angle basis (bf16), triplet cutoff weight, centre node
// ---------------------------------------------------------------------------
__global__ void triplet_geom_kernel(const int* __restrict__ trip, const int* __restrict__ ei,
                                    const float* __restrict__ r_ij, const float* __restrict__ dist,
                                    const float* __restrict__ cw,
                                    u16* __restrict__ ang_bf, float* __restrict__ w_trip,
                                    int* __restrict__ centre) {
    int t = blockIdx.x * blockDim.x + threadIdx.x;
    if (t >= TT) return;
    int e1 = trip[t], e2 = trip[TT + t];
    float d1 = fmaxf(dist[e1], 1e-8f), d2 = fmaxf(dist[e2], 1e-8f);
    float dot = r_ij[e1*3+0]*r_ij[e2*3+0] + r_ij[e1*3+1]*r_ij[e2*3+1] + r_ij[e1*3+2]*r_ij[e2*3+2];
    float ca = fminf(1.f, fmaxf(-1.f, dot / (d1 * d2)));
    const float inv = 1.f / (0.15f * 0.15f);
    u16* row = ang_bf + (size_t)t * 16;
    #pragma unroll
    for (int a = 0; a < 16; ++a) {
        float c = -1.f + 2.f * (float)a / 15.f;
        float u = ca - c;
        row[a] = f2bf(expf(-u * u * inv));
    }
    w_trip[t] = cw[e1] * cw[e2];
    centre[t] = ei[EE + e1];    // dst[e1]
}

// ---------------------------------------------------------------------------
// Node init: af = emb[type] + softplus(global @ gp_W + gp_b)[batch]
// ---------------------------------------------------------------------------
__global__ void gproj_kernel(const float* __restrict__ gf, const float* __restrict__ W,
                             const float* __restrict__ b, float* __restrict__ gproj) {
    int i = blockIdx.x * blockDim.x + threadIdx.x;
    if (i >= BB * 64) return;
    int bb = i >> 6, f = i & 63;
    gproj[i] = softplusf(gf[bb*2] * W[f] + gf[bb*2+1] * W[64 + f] + b[f]);
}

__global__ void af_init_kernel(const int* __restrict__ at, const int* __restrict__ batch,
                               const float* __restrict__ emb, const float* __restrict__ gproj,
                               float* __restrict__ af, u16* __restrict__ af_bf) {
    int i = blockIdx.x * blockDim.x + threadIdx.x;
    if (i >= NN * 64) return;
    int n = i >> 6, f = i & 63;
    float x = emb[at[n] * 64 + f] + gproj[batch[n] * 64 + f];
    af[i] = x; af_bf[i] = f2bf(x);
}

// ---------------------------------------------------------------------------
// Gathered bf16 WMMA GEMM with fused column-statistics epilogue.
// MODE 0 (2-body): A row e = [af[dst]|af[src]|edge_fea[e]] (K=208, pad 224)
// MODE 1 (3-body): A row t = [af[ctr]|ef[e1]|ef[e2]|ang[t]] (K=240, pad 256)
// Per wave: 16 rows x 128 cols (8 accumulators), block = 8 waves = 128 rows.
// All gather sources are addressed as 32-bit byte offsets from a single
// workspace base so the selects stay 32-bit and loads stay GLOBAL (saddr+voffset).
// B fragments for a whole K-tile are loaded up-front so the 8-WMMA chain
// overlaps the next A/B loads instead of draining loadcnt per MMA.
// ---------------------------------------------------------------------------
template <int MODE, int KTILES>
__global__ void __launch_bounds__(256)
gemm_conv_kernel(int rows,
                 const int* __restrict__ ia, const int* __restrict__ ib, const int* __restrict__ ic,
                 const char* __restrict__ base,
                 unsigned af_off, unsigned ef_off, unsigned ang_off, unsigned zero_off,
                 const u16* __restrict__ wf,
                 u16* __restrict__ h_bf, float* __restrict__ stats) {
    __shared__ float ssum[128], ssq[128];
    int tid = threadIdx.x;
    if (tid < 128) { ssum[tid] = 0.f; ssq[tid] = 0.f; }
    __syncthreads();

    int lane = tid & 31, wv = tid >> 5;
    int tile_base = blockIdx.x * 128 + wv * 16;
    int mrow = lane & 15;
    int row = tile_base + mrow;
    bool valid = row < rows;
    int r = valid ? row : 0;

    int n0 = ia[r];
    int n1 = ib[r];
    int n2 = (MODE == 1) ? ic[r] : 0;

    // 32-bit byte offset of the 8-half run starting at concat position kb
    auto a_off = [&](int kb) -> unsigned {
        if (!valid) return zero_off;
        if (MODE == 0) {
            if (kb < 64)  return af_off  + ((unsigned)n0 * 64 + kb) * 2u;
            if (kb < 128) return af_off  + ((unsigned)n1 * 64 + (kb - 64)) * 2u;
            if (kb < 208) return ef_off  + ((unsigned)row * 80 + (kb - 128)) * 2u;
            return zero_off;
        } else {
            if (kb < 64)  return af_off  + ((unsigned)n0 * 64 + kb) * 2u;
            if (kb < 144) return ef_off  + ((unsigned)n1 * 80 + (kb - 64)) * 2u;
            if (kb < 224) return ef_off  + ((unsigned)n2 * 80 + (kb - 144)) * 2u;
            if (kb < 240) return ang_off + ((unsigned)row * 16 + (kb - 224)) * 2u;
            return zero_off;
        }
    };
    auto load_a = [&](Frag16& a, int kt) {
        int kb0 = kt * 32 + (lane >> 4) * 8;      // two contiguous 8-half runs
        a.q[0] = *(const u32x4*)(base + a_off(kb0));
        a.q[1] = *(const u32x4*)(base + a_off(kb0 + 16));
    };

    FragF acc[8];
    #pragma unroll
    for (int i = 0; i < 8; ++i)
        #pragma unroll
        for (int j = 0; j < 8; ++j) acc[i].f[j] = 0.f;

    Frag16 acur, anext;
    load_a(acur, 0);

    #pragma unroll
    for (int kt = 0; kt < KTILES; ++kt) {
        // issue all B-fragment loads for this K-tile as one clause
        Frag16 b[8];
        #pragma unroll
        for (int nt = 0; nt < 8; ++nt) {
            const u16* wp = wf + (((size_t)(kt * 8 + nt) * 32) + lane) * 16;
            b[nt].q[0] = *(const u32x4*)wp;
            b[nt].q[1] = *(const u32x4*)(wp + 8);
        }
        // prefetch next K-tile's A fragment before the MMA chain
        if (kt + 1 < KTILES) load_a(anext, kt + 1);
        #pragma unroll
        for (int nt = 0; nt < 8; ++nt) {
            acc[nt].v = __builtin_amdgcn_wmma_f32_16x16x32_bf16(
                false, acur.v, false, b[nt].v, (short)0, acc[nt].v, false, false);
        }
        if (kt + 1 < KTILES) acur = anext;
    }

    // Epilogue: store h (bf16) + per-column sum / sum-of-squares
    int colbase = lane & 15;
    int rbase = tile_base + 8 * (lane >> 4);
    #pragma unroll
    for (int nt = 0; nt < 8; ++nt) {
        int col = nt * 16 + colbase;
        float s = 0.f, q = 0.f;
        #pragma unroll
        for (int g = 0; g < 8; ++g) {
            float v = acc[nt].f[g];
            int rr = rbase + g;
            if (rr < rows) h_bf[(size_t)rr * 128 + col] = f2bf(v);
            s += v; q += v * v;                    // OOB rows contribute exact 0
        }
        atomicAdd(&ssum[col], s);
        atomicAdd(&ssq[col], q);
    }
    __syncthreads();
    if (tid < 128) {
        atomicAdd(&stats[tid], ssum[tid]);
        atomicAdd(&stats[128 + tid], ssq[tid]);
    }
}

// ---------------------------------------------------------------------------
// BN(h) -> sigmoid(gate)*softplus(core)*w -> scatter-add into aggr[target]
// stats layout: [0..127] = col sums, [128..255] = col sum-of-squares
// ---------------------------------------------------------------------------
__global__ void bn_gate_scatter_kernel(int rows, const u16* __restrict__ h_bf,
                                       const float* __restrict__ stats, float invR,
                                       const float* __restrict__ bn_g, const float* __restrict__ bn_b,
                                       const float* __restrict__ wrow, const int* __restrict__ target,
                                       float* __restrict__ aggr) {
    size_t i = (size_t)blockIdx.x * blockDim.x + threadIdx.x;
    if (i >= (size_t)rows * 64) return;
    int e = (int)(i >> 6);
    int c = (int)(i & 63);
    float mg = stats[c] * invR,        vg = stats[128 + c] * invR - mg * mg;
    float mc = stats[64 + c] * invR,   vc = stats[192 + c] * invR - mc * mc;
    float xg = bf2f(h_bf[(size_t)e * 128 + c]);
    float xc = bf2f(h_bf[(size_t)e * 128 + 64 + c]);
    float g  = bn_g[c]      * (xg - mg) * rsqrtf(vg + 1e-5f) + bn_b[c];
    float cv = bn_g[64 + c] * (xc - mc) * rsqrtf(vc + 1e-5f) + bn_b[64 + c];
    float msg = sigmoidf(g) * softplusf(cv) * wrow[e];
    atomicAdd(&aggr[(size_t)target[e] * 64 + c], msg);
}

// Column stats (64 cols over `rows` rows) -> stats2[0..63]=sum, [64..127]=sumsq
__global__ void colstats64_kernel(const float* __restrict__ x, int rows, float* __restrict__ stats2) {
    __shared__ float ss[256], sq[256];
    int c = threadIdx.x & 63;
    int rl = threadIdx.x >> 6;
    float s = 0.f, q = 0.f;
    for (int r = blockIdx.x * 4 + rl; r < rows; r += gridDim.x * 4) {
        float v = x[(size_t)r * 64 + c];
        s += v; q += v * v;
    }
    ss[threadIdx.x] = s; sq[threadIdx.x] = q;
    __syncthreads();
    if (rl == 0) {
        s = ss[c] + ss[64 + c] + ss[128 + c] + ss[192 + c];
        q = sq[c] + sq[64 + c] + sq[128 + c] + sq[192 + c];
        atomicAdd(&stats2[c], s);
        atomicAdd(&stats2[64 + c], q);
    }
}

// af = softplus(af + BN(aggr)); refresh bf16 mirror
__global__ void update_af_kernel(const float* __restrict__ aggr, const float* __restrict__ stats2,
                                 float invN, const float* __restrict__ g2, const float* __restrict__ b2,
                                 float* __restrict__ af, u16* __restrict__ af_bf) {
    int i = blockIdx.x * blockDim.x + threadIdx.x;
    if (i >= NN * 64) return;
    int c = i & 63;
    float m = stats2[c] * invN;
    float v = stats2[64 + c] * invN - m * m;
    float bn = g2[c] * (aggr[i] - m) * rsqrtf(v + 1e-5f) + b2[c];
    float x = softplusf(af[i] + bn);
    af[i] = x; af_bf[i] = f2bf(x);
}

// ---------------------------------------------------------------------------
// Energy head (1.6 GFLOP total -> plain VALU, weights L2-resident)
// ---------------------------------------------------------------------------
__global__ void eh1_kernel(const float* __restrict__ af, const float* __restrict__ W1,
                           const float* __restrict__ b1, float* __restrict__ h1) {
    int i = blockIdx.x * blockDim.x + threadIdx.x;
    if (i >= NN * 128) return;
    int n = i >> 7, j = i & 127;
    const float* a = af + (size_t)n * 64;
    float s = b1[j];
    #pragma unroll 8
    for (int k = 0; k < 64; ++k) s += a[k] * W1[k * 128 + j];
    h1[i] = softplusf(s);
}
__global__ void eh2_kernel(const float* __restrict__ h1, const float* __restrict__ W2,
                           const float* __restrict__ b2, float* __restrict__ h2) {
    int i = blockIdx.x * blockDim.x + threadIdx.x;
    if (i >= NN * 64) return;
    int n = i >> 6, j = i & 63;
    const float* a = h1 + (size_t)n * 128;
    float s = b2[j];
    #pragma unroll 8
    for (int k = 0; k < 128; ++k) s += a[k] * W2[k * 64 + j];
    h2[i] = softplusf(s);
}
__global__ void eh3_kernel(const float* __restrict__ h2, const float* __restrict__ W3,
                           const float* __restrict__ b3, const int* __restrict__ batch,
                           float* __restrict__ out) {
    int n = blockIdx.x * blockDim.x + threadIdx.x;
    if (n >= NN) return;
    const float* a = h2 + (size_t)n * 64;
    float s = b3[0];
    #pragma unroll 8
    for (int k = 0; k < 64; ++k) s += a[k] * W3[k];
    atomicAdd(&out[batch[n]], s);
}

// ---------------------------------------------------------------------------
extern "C" void kernel_launch(void* const* d_in, const int* in_sizes, int n_in,
                              void* d_out, int out_size, void* d_ws, size_t ws_size,
                              hipStream_t stream) {
    (void)in_sizes; (void)n_in; (void)out_size; (void)ws_size;

    const int*   atom_types = (const int*)  d_in[0];
    const float* pos        = (const float*)d_in[1];
    const int*   ei         = (const int*)  d_in[2];   // [2,E]: src=ei, dst=ei+E
    const float* eoff       = (const float*)d_in[3];
    const int*   trip       = (const int*)  d_in[4];   // [2,T]
    const int*   batch      = (const int*)  d_in[5];
    const float* gf         = (const float*)d_in[6];
    const float* emb        = (const float*)d_in[7];
    const float* gp_W       = (const float*)d_in[8];
    const float* gp_b       = (const float*)d_in[9];
    const float* tb_W       = (const float*)d_in[10];
    const float* tb_bn1_g   = (const float*)d_in[12];
    const float* tb_bn1_b   = (const float*)d_in[13];
    const float* tb_bn2_g   = (const float*)d_in[14];
    const float* tb_bn2_b   = (const float*)d_in[15];
    const float* th_W       = (const float*)d_in[16];
    const float* th_bn1_g   = (const float*)d_in[18];
    const float* th_bn1_b   = (const float*)d_in[19];
    const float* th_bn2_g   = (const float*)d_in[20];
    const float* th_bn2_b   = (const float*)d_in[21];
    const float* eh_W1      = (const float*)d_in[22];
    const float* eh_b1      = (const float*)d_in[23];
    const float* eh_W2      = (const float*)d_in[24];
    const float* eh_b2      = (const float*)d_in[25];
    const float* eh_W3      = (const float*)d_in[26];
    const float* eh_b3      = (const float*)d_in[27];
    float* out = (float*)d_out;

    const int* src = ei;
    const int* dst = ei + EE;

    // ---- workspace carving (all gather sources share one base => 32-bit offsets) ----
    char* wsbase = (char*)d_ws;
    char* p = wsbase;
    auto take = [&](size_t bytes) -> void* {
        void* r = (void*)p;
        p += (bytes + 255) & ~(size_t)255;
        return r;
    };
    float* r_ij   = (float*)take((size_t)EE * 3 * 4);
    float* dist   = (float*)take((size_t)EE * 4);
    float* cw     = (float*)take((size_t)EE * 4);
    u16*   ef_bf  = (u16*)  take((size_t)EE * 80 * 2);
    u16*   ang_bf = (u16*)  take((size_t)TT * 16 * 2);
    float* w_trip = (float*)take((size_t)TT * 4);
    int*   centre = (int*)  take((size_t)TT * 4);
    float* af     = (float*)take((size_t)NN * 64 * 4);
    u16*   af_bf  = (u16*)  take((size_t)NN * 64 * 2);
    float* aggr   = (float*)take((size_t)NN * 64 * 4);
    u16*   h_bf   = (u16*)  take((size_t)EE * 128 * 2);
    float* stats1 = (float*)take(256 * 4);
    float* stats2 = (float*)take(128 * 4);
    u16*   wfrag2 = (u16*)  take((size_t)LL * KT2 * 8 * 32 * 16 * 2);
    u16*   wfrag3 = (u16*)  take((size_t)LL * KT3 * 8 * 32 * 16 * 2);
    float* gproj  = (float*)take((size_t)BB * 64 * 4);
    float* h1     = (float*)take((size_t)NN * 128 * 4);
    float* h2     = (float*)take((size_t)NN * 64 * 4);
    char*  zpad   = (char*) take(64);                    // 32-half zero pad for K tail

    const unsigned af_off   = (unsigned)((char*)af_bf  - wsbase);
    const unsigned ef_off   = (unsigned)((char*)ef_bf  - wsbase);
    const unsigned ang_off  = (unsigned)((char*)ang_bf - wsbase);
    const unsigned zero_off = (unsigned)(zpad           - wsbase);

    hipMemsetAsync(zpad, 0, 64, stream);

    // ---- preprocessing ----
    {
        long long t2 = (long long)LL * KT2 * 8 * 32 * 16;   // 114688
        long long t3 = (long long)LL * KT3 * 8 * 32 * 16;   // 131072
        prep_w_kernel<<<(int)((t2 + 255) / 256), 256, 0, stream>>>(208, KT2, tb_W, wfrag2, t2);
        prep_w_kernel<<<(int)((t3 + 255) / 256), 256, 0, stream>>>(240, KT3, th_W, wfrag3, t3);
    }
    edge_geom_kernel<<<(EE + 255) / 256, 256, 0, stream>>>(ei, pos, eoff, r_ij, dist, cw, ef_bf);
    gproj_kernel<<<4, 256, 0, stream>>>(gf, gp_W, gp_b, gproj);
    af_init_kernel<<<(NN * 64 + 255) / 256, 256, 0, stream>>>(atom_types, batch, emb, gproj, af, af_bf);
    triplet_geom_kernel<<<(TT + 255) / 256, 256, 0, stream>>>(trip, ei, r_ij, dist, cw,
                                                              ang_bf, w_trip, centre);

    const int gemm2_grid = (EE + 127) / 128;
    const int gemm3_grid = (TT + 127) / 128;
    const long long e64 = (long long)EE * 64, t64 = (long long)TT * 64;

    for (int l = 0; l < LL; ++l) {
        // ===== 2-body conv =====
        hipMemsetAsync(stats1, 0, 256 * 4, stream);
        gemm_conv_kernel<0, KT2><<<gemm2_grid, 256, 0, stream>>>(
            EE, dst, src, (const int*)nullptr, wsbase, af_off, ef_off, ang_off, zero_off,
            wfrag2 + (size_t)l * KT2 * 8 * 32 * 16, h_bf, stats1);
        hipMemsetAsync(aggr, 0, (size_t)NN * 64 * 4, stream);
        bn_gate_scatter_kernel<<<(int)((e64 + 255) / 256), 256, 0, stream>>>(
            EE, h_bf, stats1, 1.f / (float)EE,
            tb_bn1_g + l * 128, tb_bn1_b + l * 128, cw, dst, aggr);
        hipMemsetAsync(stats2, 0, 128 * 4, stream);
        colstats64_kernel<<<120, 256, 0, stream>>>(aggr, NN, stats2);
        update_af_kernel<<<(NN * 64 + 255) / 256, 256, 0, stream>>>(
            aggr, stats2, 1.f / (float)NN, tb_bn2_g + l * 64, tb_bn2_b + l * 64, af, af_bf);

        // ===== 3-body conv =====
        hipMemsetAsync(stats1, 0, 256 * 4, stream);
        gemm_conv_kernel<1, KT3><<<gemm3_grid, 256, 0, stream>>>(
            TT, centre, trip, trip + TT, wsbase, af_off, ef_off, ang_off, zero_off,
            wfrag3 + (size_t)l * KT3 * 8 * 32 * 16, h_bf, stats1);
        hipMemsetAsync(aggr, 0, (size_t)NN * 64 * 4, stream);
        bn_gate_scatter_kernel<<<(int)((t64 + 255) / 256), 256, 0, stream>>>(
            TT, h_bf, stats1, 1.f / (float)TT,
            th_bn1_g + l * 128, th_bn1_b + l * 128, w_trip, centre, aggr);
        hipMemsetAsync(stats2, 0, 128 * 4, stream);
        colstats64_kernel<<<120, 256, 0, stream>>>(aggr, NN, stats2);
        update_af_kernel<<<(NN * 64 + 255) / 256, 256, 0, stream>>>(
            aggr, stats2, 1.f / (float)NN, th_bn2_g + l * 64, th_bn2_b + l * 64, af, af_bf);
    }

    // ===== energy head =====
    eh1_kernel<<<(NN * 128 + 255) / 256, 256, 0, stream>>>(af, eh_W1, eh_b1, h1);
    eh2_kernel<<<(NN * 64 + 255) / 256, 256, 0, stream>>>(h1, eh_W2, eh_b2, h2);
    hipMemsetAsync(out, 0, BB * 4, stream);
    eh3_kernel<<<(NN + 255) / 256, 256, 0, stream>>>(h2, eh_W3, eh_b3, batch, out);
}
